// ReweightVLAD_53206054863273
// MI455X (gfx1250) — matrix operand order repaired and checked by reading, not compiled, not run on hardware
//
#include <hip/hip_runtime.h>
#include <hip/hip_bf16.h>

// ---------------------------------------------------------------------------
// ReweightVLAD on MI455X (gfx1250, wave32, WMMA, TDM/async-LDS).
//  K0 pack_a      : pre-pack conv/cluster weights to bf16 in WMMA A-fragment order
//  K1 ln_upsample : LayerNorm(7x7) + bilinear 7->13, store h13 as [n][pix][cin] bf16
//  K2 xnorm       : per-pixel channel L2 norm; xn in [pix][cin] and [cin][pix] bf16
//  K3 conv_wmma   : c3/c5/c7 implicit GEMM; h13[n] (173KB) staged in LDS via TDM
//  K4 crn_tail    : 13->7 separable reductions + 84->1 accumulation -> cw
//  K5 logits_mask : clusters GEMM (WMMA) + softmax + *cw; pack mask as A-fragments
//  K6 vlad_wmma   : mask(64x64) x xn(64x512) WMMA - centroid*msum -> d_out
//  K7 norm        : intra-cluster L2 norm + global L2 norm
// ---------------------------------------------------------------------------

typedef __bf16 bf16_t;
typedef __bf16 v16bf __attribute__((ext_vector_type(16)));
typedef float  v8f   __attribute__((ext_vector_type(8)));
typedef unsigned int v4u __attribute__((ext_vector_type(4)));
typedef int v8i __attribute__((ext_vector_type(8)));
typedef int v4i __attribute__((ext_vector_type(4)));

#define NB      512
#define CCH     512
#define KCL     64
#define HW7     49
#define P13     169
#define OC_ALL  84

#if defined(__has_builtin)
# if __has_builtin(__builtin_amdgcn_global_load_async_to_lds_b128) && \
     __has_builtin(__builtin_amdgcn_s_wait_asynccnt)
#  define HAVE_ASYNC_LDS 1
# endif
# if __has_builtin(__builtin_amdgcn_tensor_load_to_lds) && \
     __has_builtin(__builtin_amdgcn_s_wait_tensorcnt)
#  define HAVE_TDM 1
# endif
#endif

// Generic async stage: global -> LDS, 16B per lane per issue, ASYNCcnt tracked.
// Builtin signature (probe: clang diagnostic): (v4i AS1*, v4i AS3*, imm, imm).
// Caller must __syncthreads() afterwards.
__device__ __forceinline__ void stage_to_lds(const char* __restrict__ g,
                                             char* __restrict__ l,
                                             int bytes, int t) {
#if defined(HAVE_ASYNC_LDS)
    typedef __attribute__((address_space(1))) v4i* gvp;
    typedef __attribute__((address_space(3))) v4i* lvp;
    for (int i = t * 16; i < bytes; i += 256 * 16) {
        __builtin_amdgcn_global_load_async_to_lds_b128(
            (gvp)(g + i), (lvp)(l + i), 0, 0);
    }
    __builtin_amdgcn_s_wait_asynccnt(0);
#else
    const uint4* s = (const uint4*)g;
    uint4* d = (uint4*)l;
    for (int i = t; i < bytes / 16; i += 256) d[i] = s[i];
#endif
}

// A-fragment k order (contiguous, matching B loads): element i of lane half hf
// holds K = 16*hf + i.
__device__ __forceinline__ int a_klocal(int i, int hf) {
    return hf * 16 + i;
}

// ---------------------------------------------------------------------------
// K0: pack weights (OC, 512, KH, KW) f32 -> bf16 A-fragments.
// Global k ordering: k = (dy*KW+dx)*512 + cin.
// ---------------------------------------------------------------------------
__global__ void pack_a_kernel(const float* __restrict__ w, bf16_t* __restrict__ dst,
                              int OC, int KPOS, int m_tiles, int KC) {
    int t = blockIdx.x * blockDim.x + threadIdx.x;
    int total = m_tiles * KC * 32;
    if (t >= total) return;
    int mt  = t / (KC * 32);
    int rem = t - mt * (KC * 32);
    int kc  = rem >> 5;
    int ln  = rem & 31;
    int m   = mt * 16 + (ln & 15);
    int hf  = ln >> 4;
    bf16_t* o = dst + (size_t)t * 16;
    for (int i = 0; i < 16; ++i) {
        int k   = kc * 32 + a_klocal(i, hf);
        int pos = k >> 9;
        int cin = k & 511;
        float v = 0.f;
        if (m < OC) v = w[(size_t)(m * CCH + cin) * KPOS + pos];
        o[i] = (bf16_t)v;
    }
}

// ---------------------------------------------------------------------------
// K1: LayerNorm(7x7) per (n,c) + bilinear 7->13 (align_corners). One wave per
// (n,c). Output: h13[n][pix][cin] bf16 (cin contiguous, ready for B-fragments).
// ---------------------------------------------------------------------------
__global__ void ln_upsample_kernel(const float* __restrict__ x,
                                   const float* __restrict__ lnw,
                                   const float* __restrict__ lnb,
                                   bf16_t* __restrict__ h13) {
    __shared__ float sh[8][HW7];
    int wv   = threadIdx.x >> 5;
    int lane = threadIdx.x & 31;
    int p = blockIdx.x * 8 + wv;
    int n = p >> 9;
    const float* xs = x + (size_t)p * HW7;
    float s = 0.f, ss = 0.f;
    for (int j = lane; j < HW7; j += 32) {
        float v = xs[j];
        s += v; ss += v * v;
    }
    for (int off = 16; off > 0; off >>= 1) {
        s  += __shfl_xor(s,  off, 32);
        ss += __shfl_xor(ss, off, 32);
    }
    float mu   = s * (1.f / 49.f);
    float var  = ss * (1.f / 49.f) - mu * mu;
    float rstd = rsqrtf(var + 1e-5f);
    for (int j = lane; j < HW7; j += 32)
        sh[wv][j] = (xs[j] - mu) * rstd * lnw[j] + lnb[j];
    __syncthreads();
    int c = p & 511;
    bf16_t* o = h13 + (size_t)n * P13 * CCH + c;
    for (int j = lane; j < P13; j += 32) {
        int oy = j / 13, ox = j - oy * 13;
        int iy0 = oy >> 1, ix0 = ox >> 1;
        float wy = (oy & 1) ? 0.5f : 0.f;
        float wx = (ox & 1) ? 0.5f : 0.f;
        int iy1 = min(iy0 + 1, 6), ix1 = min(ix0 + 1, 6);
        float r0 = sh[wv][iy0 * 7 + ix0] * (1.f - wx) + sh[wv][iy0 * 7 + ix1] * wx;
        float r1 = sh[wv][iy1 * 7 + ix0] * (1.f - wx) + sh[wv][iy1 * 7 + ix1] * wx;
        o[(size_t)j * CCH] = (bf16_t)(r0 * (1.f - wy) + r1 * wy);
    }
}

// ---------------------------------------------------------------------------
// K2: per-pixel channel L2 normalization; xn_pc[n][pix64][cin] (logits B) and
// xn_cp[n][cin][pix64] (VLAD B), both bf16, pixel pad 49->64 with zeros.
// ---------------------------------------------------------------------------
__global__ void xnorm_kernel(const float* __restrict__ x,
                             bf16_t* __restrict__ xn_pc,
                             bf16_t* __restrict__ xn_cp) {
    extern __shared__ char smem[];
    float* xb = (float*)smem;                   // 512*49 f32
    __shared__ float ssum[4][64];
    __shared__ float sinv[64];
    int n = blockIdx.x, t = threadIdx.x;
    stage_to_lds((const char*)(x + (size_t)n * CCH * HW7), smem, CCH * HW7 * 4, t);
    __syncthreads();
    int pix = t & 63, g = t >> 6;
    float acc = 0.f;
    if (pix < HW7)
        for (int c = g * 128; c < g * 128 + 128; ++c) {
            float v = xb[c * HW7 + pix];
            acc += v * v;
        }
    ssum[g][pix] = acc;
    __syncthreads();
    if (t < 64) {
        float sall = ssum[0][t] + ssum[1][t] + ssum[2][t] + ssum[3][t];
        float inv  = 1.f / fmaxf(sqrtf(sall), 1e-12f);
        sinv[t] = (t < HW7) ? inv : 0.f;
    }
    __syncthreads();
    bf16_t* ocp = xn_cp + (size_t)n * CCH * 64;
    bf16_t* opc = xn_pc + (size_t)n * 64 * CCH;
    for (int i = t; i < CCH * 64; i += 256) {
        int c = i >> 6, px = i & 63;
        float v = (px < HW7) ? xb[c * HW7 + px] * sinv[px] : 0.f;
        ocp[i] = (bf16_t)v;
    }
    for (int i = t; i < 64 * CCH; i += 256) {
        int px = i >> 9, c = i & 511;
        float v = (px < HW7) ? xb[c * HW7 + px] * sinv[px] : 0.f;
        opc[i] = (bf16_t)v;
    }
}

// ---------------------------------------------------------------------------
// K3: implicit-GEMM conv via WMMA bf16 16x16x32. One block per n; h13[n]
// (169x512 bf16 = 173KB of the 320KB WGP LDS) staged via the Tensor Data
// Mover (one TENSOR_LOAD_TO_LDS from wave 0; 2-D D#: 169 rows x 512 bf16).
// Inner loop: per kernel-position, 16 chunks of 32 channels; each chunk is
// 2x ds_load_b128 (B) + 2x global_load_b128 (prepacked A, L2-resident) + wmma.
// ---------------------------------------------------------------------------
__global__ void conv_wmma_kernel(const bf16_t* __restrict__ h13,
                                 const bf16_t* __restrict__ wpack,
                                 const float* __restrict__ b3,
                                 const float* __restrict__ b5,
                                 const float* __restrict__ b7,
                                 float* __restrict__ hconv) {
    extern __shared__ char smem[];
    bf16_t* hb = (bf16_t*)smem;                 // [169][512] bf16
    int n = blockIdx.x, t = threadIdx.x;

#if defined(HAVE_TDM)
    if (t < 32) {   // one TDM issue for the whole block (TDM ignores EXEC)
        unsigned long long ga =
            (unsigned long long)(size_t)(h13 + (size_t)n * P13 * CCH);
        // D# group0: count=1, lds_addr=0 (dynamic-LDS base), global_addr, type=2
        v4u g0 = { 1u, 0u, (unsigned)(ga & 0xffffffffull),
                   (unsigned)((ga >> 32) & 0x1ffffffull) | 0x80000000u };
        // D# group1: data_size=2B; tensor_dim0=512, tensor_dim1=169;
        // tile_dim0=512, tile_dim1=169; tensor_dim0_stride=512
        v8i g1 = { (int)(1u << 16), (int)(512u << 16), (int)(169u << 16),
                   (int)(512u << 16), 169, 512, 0, 0 };
        v4i gz = { 0, 0, 0, 0 };
#if defined(__clang_major__) && (__clang_major__ >= 23)
        v8i gz8 = { 0, 0, 0, 0, 0, 0, 0, 0 };
        __builtin_amdgcn_tensor_load_to_lds(g0, g1, gz, gz, gz8, 0);
#else
        __builtin_amdgcn_tensor_load_to_lds(g0, g1, gz, gz, 0);
#endif
        __builtin_amdgcn_s_wait_tensorcnt(0);
    }
#else
    stage_to_lds((const char*)(h13 + (size_t)n * P13 * CCH), smem, P13 * CCH * 2, t);
#endif
    __syncthreads();

    const int KWs[3]   = {3, 5, 7};
    const int PADs[3]  = {1, 2, 3};
    const int KPOSs[3] = {9, 25, 49};
    const int WOFF[3]  = {0, 147456, 557056};   // element offsets into wpack
    const int OCB[3]   = {0, 32, 64};
    const int OCs[3]   = {32, 32, 20};

    int wv = t >> 5, lane = t & 31, hf = lane >> 4;
    v16bf zf;
    for (int i = 0; i < 16; ++i) zf[i] = (bf16_t)0.f;

    for (int tile = wv; tile < 66; tile += 8) {
        int conv = (tile < 22) ? 0 : (tile < 44 ? 1 : 2);
        int loc  = tile - conv * 22;
        int mt   = loc / 11, nt = loc - mt * 11;
        int KW = KWs[conv], pad = PADs[conv], KPOS = KPOSs[conv];
        const bf16_t* ap = wpack + (size_t)WOFF[conv]
                         + (size_t)mt * (KPOS * 16) * 32 * 16;

        int p  = nt * 16 + (lane & 15);
        bool pv = p < P13;
        int pp = pv ? p : 0;
        int oy = pp / 13, ox = pp - oy * 13;

        v8f acc;
        for (int i = 0; i < 8; ++i) acc[i] = 0.f;

        for (int pos = 0; pos < KPOS; ++pos) {
            int dy = pos / KW, dx = pos - dy * KW;
            int iy = oy + dy - pad, ix = ox + dx - pad;
            bool valid = pv & (iy >= 0) & (iy < 13) & (ix >= 0) & (ix < 13);
            const bf16_t* bbase =
                hb + (valid ? ((iy * 13 + ix) << 9) + (hf << 4) : 0);
            const bf16_t* al = ap + (((size_t)pos * 16) * 32 + lane) * 16;
            __builtin_prefetch(al + 16 * 32 * 16, 0, 1);   // next position's A
#pragma unroll 4
            for (int q = 0; q < 16; ++q) {                 // 16 chunks of 32 ch
                v16bf bv = *(const v16bf*)(bbase + q * 32);
                v16bf bfrag = valid ? bv : zf;              // zero-pad halo
                v16bf af = *(const v16bf*)(al + (size_t)q * 32 * 16);
                acc = __builtin_amdgcn_wmma_f32_16x16x32_bf16(
                          false, af, false, bfrag, (short)0, acc, false, false);
            }
        }

        const float* bp = (conv == 0) ? b3 : (conv == 1 ? b5 : b7);
        int OC = OCs[conv];
        for (int v = 0; v < 8; ++v) {
            int m = mt * 16 + v + (hf << 3);
            if (m < OC && pv)
                hconv[((size_t)n * OC_ALL + OCB[conv] + m) * P13 + p] = acc[v] + bp[m];
        }
    }
}

// ---------------------------------------------------------------------------
// K4: 13->7 width/height reductions + 84->1 accumulation -> cw[n][49]
// (flattening l = ow*7 + oh, matching the reference transpose).
// ---------------------------------------------------------------------------
__global__ void crn_tail_kernel(const float* __restrict__ hconv,
                                const float* __restrict__ dww, const float* __restrict__ dwb,
                                const float* __restrict__ dhw, const float* __restrict__ dhb,
                                const float* __restrict__ accw, const float* __restrict__ accb,
                                float* __restrict__ cw) {
    extern __shared__ char smem[];
    float* hc = (float*)smem;                    // 84*169
    float* t1 = hc + OC_ALL * P13;               // 84*13*7
    int n = blockIdx.x, t = threadIdx.x;
    stage_to_lds((const char*)(hconv + (size_t)n * OC_ALL * P13), smem,
                 OC_ALL * P13 * 4, t);
    __syncthreads();
    for (int i = t; i < OC_ALL * 13 * 7; i += 256) {
        int c = i / 91, r = i - c * 91;
        int h = r / 7, ow = r - h * 7;
        float sum = dwb[ow];
        for (int w = 0; w < 13; ++w) sum += hc[c * P13 + h * 13 + w] * dww[ow * 13 + w];
        t1[c * 91 + h * 7 + ow] = sum;
    }
    __syncthreads();
    float* t2 = hc;                              // reuse; [c][ow*7+oh]
    for (int i = t; i < OC_ALL * HW7; i += 256) {
        int c = i / HW7, r = i - c * HW7;
        int ow = r / 7, oh = r - ow * 7;
        float sum = dhb[oh];
        for (int h = 0; h < 13; ++h) sum += t1[c * 91 + h * 7 + ow] * dhw[oh * 13 + h];
        t2[c * HW7 + r] = sum;
    }
    __syncthreads();
    if (t < HW7) {
        float sum = accb[0];
        for (int c = 0; c < OC_ALL; ++c) sum += t2[c * HW7 + t] * accw[c];
        cw[n * 64 + t] = sum;
    }
}

// ---------------------------------------------------------------------------
// K5: logits = conv_w(64x512) @ xn(512x64pix) WMMA + bias; softmax over
// clusters; mask = sa*cw; msum; re-pack mask into WMMA A-fragment order.
// ---------------------------------------------------------------------------
__global__ void logits_mask_kernel(const bf16_t* __restrict__ xn_pc,
                                   const bf16_t* __restrict__ wcl,
                                   const float* __restrict__ convb,
                                   const float* __restrict__ cw,
                                   float* __restrict__ msum,
                                   bf16_t* __restrict__ amask) {
    extern __shared__ char smem[];
    bf16_t* xs = (bf16_t*)smem;                       // 64*512 bf16
    float* lg = (float*)(smem + 64 * CCH * 2);        // 64*64 logits
    float* mk = lg + 64 * 64;                         // 64*64 mask
    int n = blockIdx.x, t = threadIdx.x;
    stage_to_lds((const char*)(xn_pc + (size_t)n * 64 * CCH), smem, 64 * CCH * 2, t);
    __syncthreads();
    int wv = t >> 5, lane = t & 31, hf = lane >> 4;
    for (int tile = wv; tile < 16; tile += 8) {
        int mt = tile >> 2, nt = tile & 3;
        int pix = nt * 16 + (lane & 15);
        v8f acc;
        for (int i = 0; i < 8; ++i) acc[i] = 0.f;
        for (int kc = 0; kc < 16; ++kc) {
            v16bf af = *(const v16bf*)(wcl + ((size_t)(mt * 16 + kc) * 32 + lane) * 16);
            v16bf bfrag = *(const v16bf*)(xs + (pix << 9) + (kc << 5) + (hf << 4));
            acc = __builtin_amdgcn_wmma_f32_16x16x32_bf16(
                      false, af, false, bfrag, (short)0, acc, false, false);
        }
        for (int v = 0; v < 8; ++v) {
            int m = mt * 16 + v + (hf << 3);
            lg[m * 64 + pix] = acc[v] + convb[m];
        }
    }
    __syncthreads();
    if (t < 64) {
        int pix = t;
        float mx = -1e30f;
        for (int k = 0; k < KCL; ++k) mx = fmaxf(mx, lg[k * 64 + pix]);
        float se = 0.f;
        for (int k = 0; k < KCL; ++k) se += __expf(lg[k * 64 + pix] - mx);
        float cwv = (pix < HW7) ? cw[n * 64 + pix] : 0.f;
        float sc = cwv / se;
        for (int k = 0; k < KCL; ++k)
            mk[k * 64 + pix] = (pix < HW7) ? __expf(lg[k * 64 + pix] - mx) * sc : 0.f;
    }
    __syncthreads();
    if (t < 64) {
        float s = 0.f;
        for (int p = 0; p < HW7; ++p) s += mk[t * 64 + p];
        msum[n * 64 + t] = s;
    }
    {   // pack mask A-fragments: thread t = (mt*2 + kc)*32 + lane
        int mt = t >> 6, kc = (t >> 5) & 1, ln = t & 31;
        int m = mt * 16 + (ln & 15), h2 = ln >> 4;
        bf16_t* o = amask + (size_t)n * 4096 + (size_t)t * 16;
        for (int i = 0; i < 16; ++i) {
            int kp = kc * 32 + a_klocal(i, h2);
            o[i] = (bf16_t)mk[m * 64 + kp];
        }
    }
}

// ---------------------------------------------------------------------------
// K6: vlad[n] = mask(64x64) @ xn^T(64x512) - centroids*msum  via WMMA.
// ---------------------------------------------------------------------------
__global__ void vlad_wmma_kernel(const bf16_t* __restrict__ xn_cp,
                                 const bf16_t* __restrict__ amask,
                                 const float* __restrict__ msum,
                                 const float* __restrict__ centroids,
                                 float* __restrict__ out) {
    extern __shared__ char smem[];
    bf16_t* xs = (bf16_t*)smem;                 // [512][64] bf16
    int n = blockIdx.x, t = threadIdx.x;
    stage_to_lds((const char*)(xn_cp + (size_t)n * CCH * 64), smem, CCH * 64 * 2, t);
    __syncthreads();
    int wv = t >> 5, lane = t & 31, hf = lane >> 4;
    const bf16_t* ab = amask + (size_t)n * 4096;
    for (int tile = wv; tile < 128; tile += 8) {
        int mt = tile >> 5, nt = tile & 31;
        int c = nt * 16 + (lane & 15);
        v8f acc;
        for (int i = 0; i < 8; ++i) acc[i] = 0.f;
        for (int kc = 0; kc < 2; ++kc) {
            v16bf af = *(const v16bf*)(ab + ((size_t)(mt * 2 + kc) * 32 + lane) * 16);
            v16bf bfrag = *(const v16bf*)(xs + (c << 6) + (kc << 5) + (hf << 4));
            acc = __builtin_amdgcn_wmma_f32_16x16x32_bf16(
                      false, af, false, bfrag, (short)0, acc, false, false);
        }
        for (int v = 0; v < 8; ++v) {
            int m = mt * 16 + v + (hf << 3);
            out[(size_t)n * (KCL * CCH) + (size_t)m * CCH + c] =
                acc[v] - centroids[m * CCH + c] * msum[n * 64 + m];
        }
    }
}

// ---------------------------------------------------------------------------
// K7: intra-cluster L2 normalization + global L2 normalization (in place).
// ---------------------------------------------------------------------------
__global__ void norm_kernel(float* __restrict__ out) {
    __shared__ float red[256];
    __shared__ float sval[2];
    int n = blockIdx.x, t = threadIdx.x;
    float* o = out + (size_t)n * KCL * CCH;
    if (t == 0) sval[1] = 0.f;
    __syncthreads();
    for (int k = 0; k < KCL; ++k) {
        float v1 = o[k * CCH + t];
        float v2 = o[k * CCH + t + 256];
        red[t] = v1 * v1 + v2 * v2;
        __syncthreads();
        for (int s = 128; s > 0; s >>= 1) {
            if (t < s) red[t] += red[t + s];
            __syncthreads();
        }
        if (t == 0) {
            float inv = 1.f / fmaxf(sqrtf(red[0]), 1e-12f);
            sval[0] = inv;
            sval[1] += red[0] * inv * inv;
        }
        __syncthreads();
        float inv = sval[0];
        o[k * CCH + t] = v1 * inv;
        o[k * CCH + t + 256] = v2 * inv;
        __syncthreads();
    }
    if (t == 0) sval[0] = 1.f / fmaxf(sqrtf(sval[1]), 1e-12f);
    __syncthreads();
    float g = sval[0];
    for (int i = t; i < KCL * CCH; i += 256) o[i] *= g;
}

// ---------------------------------------------------------------------------
extern "C" void kernel_launch(void* const* d_in, const int* in_sizes, int n_in,
                              void* d_out, int out_size, void* d_ws, size_t ws_size,
                              hipStream_t stream) {
    const float* x     = (const float*)d_in[0];
    const float* cent  = (const float*)d_in[1];
    const float* convw = (const float*)d_in[2];
    const float* convb = (const float*)d_in[3];
    const float* lnw   = (const float*)d_in[4];
    const float* lnb   = (const float*)d_in[5];
    const float* c3w   = (const float*)d_in[6];
    const float* c3b   = (const float*)d_in[7];
    const float* c5w   = (const float*)d_in[8];
    const float* c5b   = (const float*)d_in[9];
    const float* c7w   = (const float*)d_in[10];
    const float* c7b   = (const float*)d_in[11];
    const float* dww   = (const float*)d_in[12];
    const float* dwb   = (const float*)d_in[13];
    const float* dhw   = (const float*)d_in[14];
    const float* dhb   = (const float*)d_in[15];
    const float* accw  = (const float*)d_in[16];
    const float* accb  = (const float*)d_in[17];
    float* out = (float*)d_out;

    char* ws = (char*)d_ws;
    bf16_t* h13   = (bf16_t*)(ws + 0);               //  88,604,672 B
    bf16_t* xnpc  = (bf16_t*)(ws + 88604672ull);     //  33,554,432 B
    bf16_t* xncp  = (bf16_t*)(ws + 122159104ull);    //  33,554,432 B
    float*  hconv = (float*) (ws + 155713536ull);    //  29,073,408 B
    float*  cw    = (float*) (ws + 184786944ull);    //     131,072 B
    float*  msum  = (float*) (ws + 184918016ull);    //     131,072 B
    bf16_t* amask = (bf16_t*)(ws + 185049088ull);    //   4,194,304 B
    bf16_t* wpack = (bf16_t*)(ws + 189243392ull);    //   2,785,280 B

    pack_a_kernel<<<(2 * 144 * 32) / 256, 256, 0, stream>>>(c3w, wpack + 0,       32,  9, 2, 144);
    pack_a_kernel<<<(2 * 400 * 32) / 256, 256, 0, stream>>>(c5w, wpack + 147456,  32, 25, 2, 400);
    pack_a_kernel<<<(2 * 784 * 32) / 256, 256, 0, stream>>>(c7w, wpack + 557056,  20, 49, 2, 784);
    pack_a_kernel<<<(4 * 16 * 32) / 256, 256, 0, stream>>>(convw, wpack + 1359872, 64, 1, 4, 16);

    ln_upsample_kernel<<<(NB * CCH) / 8, 256, 0, stream>>>(x, lnw, lnb, h13);
    xnorm_kernel<<<NB, 256, CCH * HW7 * 4, stream>>>(x, xnpc, xncp);
    conv_wmma_kernel<<<NB, 256, P13 * CCH * 2, stream>>>(h13, wpack, c3b, c5b, c7b, hconv);
    crn_tail_kernel<<<NB, 256, (OC_ALL * P13 + OC_ALL * 91) * 4, stream>>>(
        hconv, dww, dwb, dhw, dhb, accw, accb, cw);
    logits_mask_kernel<<<NB, 256, 64 * CCH * 2 + 2 * 64 * 64 * 4, stream>>>(
        xnpc, wpack + 1359872, convb, cw, msum, amask);
    vlad_wmma_kernel<<<NB, 256, CCH * 64 * 2, stream>>>(xncp, amask, msum, cent, out);
    norm_kernel<<<NB, 256, 0, stream>>>(out);
}
